// MaskedHeteroGAT_23158463660139
// MI455X (gfx1250) — compile-verified, matrix operand order checked by reading, not run on hardware
//
#include <hip/hip_runtime.h>
#include <hip/hip_bf16.h>

// ---------------- problem constants ----------------
static constexpr int T  = 6;
static constexpr int NN = 50000;
static constexpr int EE = 250000;
static constexpr int DD = 256;   // feature dim
static constexpr int HH = 4;     // heads
static constexpr int CC = 64;    // channels/head (HH*CC == DD)
static constexpr int KK = 64;    // clusters

typedef __attribute__((ext_vector_type(16))) _Float16 v16h;
typedef __attribute__((ext_vector_type(8)))  _Float16 v8h;
typedef __attribute__((ext_vector_type(8)))  float    v8f;

// ---------------- helpers ----------------
__device__ __forceinline__ unsigned fenc(float x) {
    unsigned u = __float_as_uint(x);
    return (u & 0x80000000u) ? ~u : (u | 0x80000000u);
}
__device__ __forceinline__ float fdec(unsigned u) {
    return (u & 0x80000000u) ? __uint_as_float(u & 0x7fffffffu)
                             : __uint_as_float(~u);
}

// async 16B global -> LDS copy (per enabled lane), tracked by ASYNCcnt.
// GVS addressing: mem = SGPR64(base) + VGPR_I32(off) ; LDS[vdst] = mem[0:15]
__device__ __forceinline__ void async_g2l_b128(unsigned lds_byte_off,
                                               unsigned gbl_byte_off,
                                               const void* base) {
    asm volatile("global_load_async_to_lds_b128 %0, %1, %2"
                 :: "v"(lds_byte_off), "v"(gbl_byte_off), "s"(base)
                 : "memory");
}
__device__ __forceinline__ void wait_async0() {
    asm volatile("s_wait_asynccnt 0x0" ::: "memory");
}

// ---------------- utility kernels ----------------
__global__ void zero_u32(unsigned* __restrict__ p, long n) {
    long i = (long)blockIdx.x * blockDim.x + threadIdx.x;
    if (i < n) p[i] = 0u;
}

// xs1 = f16(x_pkg*nm), xd1 = f16(x_dst5*nm), xs2 = f16(relu(x_pkg))
__global__ void prep_inputs(const float* __restrict__ x_pkg,
                            const float* __restrict__ x_dst5,
                            const float* __restrict__ nmask,
                            _Float16* __restrict__ xs1,
                            _Float16* __restrict__ xd1,
                            _Float16* __restrict__ xs2, int total) {
    int i = blockIdx.x * blockDim.x + threadIdx.x;
    if (i >= total) return;
    float nm = nmask[i >> 8];
    float xp = x_pkg[i];
    xs1[i] = (_Float16)(xp * nm);
    xd1[i] = (_Float16)(x_dst5[i] * nm);
    xs2[i] = (_Float16)fmaxf(xp, 0.f);
}

// Wt[n*256+k] = f16(W[k*256+n])   (DD x DD)
__global__ void convert_w(const float* __restrict__ W, _Float16* __restrict__ Wt) {
    int i = blockIdx.x * blockDim.x + threadIdx.x;  // DD*DD threads
    int n = i >> 8, k = i & 255;
    Wt[n * DD + k] = (_Float16)W[k * DD + n];
}

__global__ void relu_to_f16(const float* __restrict__ h, _Float16* __restrict__ o, int total) {
    int i = blockIdx.x * blockDim.x + threadIdx.x;
    if (i < total) o[i] = (_Float16)fmaxf(h[i], 0.f);
}

// ---------------- WMMA GEMM: Y(N x 256) = X(N x 256, f16) @ W(256 x 256) ----
// Wt is W transposed (n-major, f16).  128x128 tile / block, 8 waves.
// Double-buffered async global->LDS staging (ASYNCcnt), WMMA compute.
static constexpr int LSTR = 40;                        // halves/row (32 + 8 pad)
static constexpr int TILE_HALVES = 128 * LSTR;         // 5120 halves
static constexpr int TILE_BYTES  = TILE_HALVES * 2;    // 10240 B
static constexpr int GEMM_LDS    = 4 * TILE_BYTES;     // A0,B0,A1,B1 = 40 KB

__global__ __launch_bounds__(256)
void gemm_wmma_f16(const _Float16* __restrict__ X,
                   const _Float16* __restrict__ Wt,
                   float* __restrict__ Y, int nrows) {
    extern __shared__ _Float16 smem[];   // [A0 | B0 | A1 | B1]

    const int tid  = threadIdx.x;
    const int wave = tid >> 5;
    const int lane = tid & 31;
    const int m0   = blockIdx.x * 128;
    const int n0   = blockIdx.y * 128;
    const int lo   = (lane >> 4) * 8;    // K sub-offset per 16-bit A/B layout

    // this thread's two 16B chunks: chunk c -> row=c>>2, part=c&3
    const int part  = tid & 3;
    const int rowC0 = tid >> 2;              // chunk tid
    const int rowC1 = (tid + 256) >> 2;      // chunk tid+256 (same part)
    const int grA0  = min(m0 + rowC0, nrows - 1);
    const int grA1  = min(m0 + rowC1, nrows - 1);
    const unsigned ldsRow0 = (unsigned)((rowC0 * LSTR + part * 8) * 2);
    const unsigned ldsRow1 = (unsigned)((rowC1 * LSTR + part * 8) * 2);

    v8f acc[8];
#pragma unroll
    for (int i = 0; i < 8; ++i) acc[i] = {};

    // issue async fills of buffer `buf` for K-offset kk (all 4 chunks)
    auto issue = [&](int buf, int kk) {
        const unsigned aBase = (unsigned)(buf * 2 * TILE_BYTES);
        const unsigned bBase = aBase + (unsigned)TILE_BYTES;
        const unsigned kByte = (unsigned)((kk + part * 8) * 2);
        async_g2l_b128(aBase + ldsRow0, (unsigned)(grA0 * (DD * 2)) + kByte, X);
        async_g2l_b128(aBase + ldsRow1, (unsigned)(grA1 * (DD * 2)) + kByte, X);
        async_g2l_b128(bBase + ldsRow0, (unsigned)((n0 + rowC0) * (DD * 2)) + kByte, Wt);
        async_g2l_b128(bBase + ldsRow1, (unsigned)((n0 + rowC1) * (DD * 2)) + kByte, Wt);
    };

    issue(0, 0);

    for (int it = 0; it < 8; ++it) {
        const int buf = it & 1;
        wait_async0();        // this wave's fills done
        __syncthreads();      // everyone's fills done -> buffer readable
        if (it + 1 < 8) issue(buf ^ 1, (it + 1) * 32);  // overlap next tile

        const _Float16* la = smem + buf * 2 * TILE_HALVES;
        const _Float16* lb = la + TILE_HALVES;

        const int am = wave * 16 + (lane & 15);
        v16h a;
        {
            v8h aLo = *(const v8h*)&la[am * LSTR + lo];
            v8h aHi = *(const v8h*)&la[am * LSTR + 16 + lo];
#pragma unroll
            for (int i = 0; i < 8; ++i) { a[i] = aLo[i]; a[8 + i] = aHi[i]; }
        }
#pragma unroll
        for (int nt = 0; nt < 8; ++nt) {
            const int bn = nt * 16 + (lane & 15);
            v16h b;
            v8h bLo = *(const v8h*)&lb[bn * LSTR + lo];
            v8h bHi = *(const v8h*)&lb[bn * LSTR + 16 + lo];
#pragma unroll
            for (int i = 0; i < 8; ++i) { b[i] = bLo[i]; b[8 + i] = bHi[i]; }
            acc[nt] = __builtin_amdgcn_wmma_f32_16x16x32_f16(
                false, a, false, b, (short)0, acc[nt], false, false);
        }
        __syncthreads();      // all waves done reading buf before it refills
    }

    // C/D layout: lane<16 -> M = r, lane>=16 -> M = 8+r ; N = lane&15
    const int rbase = m0 + wave * 16 + (lane >> 4) * 8;
    const int ncol  = n0 + (lane & 15);
#pragma unroll
    for (int nt = 0; nt < 8; ++nt) {
#pragma unroll
        for (int r = 0; r < 8; ++r) {
            int row = rbase + r;
            if (row < nrows) Y[(long)row * DD + ncol + nt * 16] = acc[nt][r];
        }
    }
}

// ---------------- edge kernels (one thread per (edge, head)) ----------------
__global__ void edge_scores(const float* __restrict__ hs, const float* __restrict__ hd,
                            const int* __restrict__ src, const int* __restrict__ dst,
                            const float* __restrict__ avec,
                            float* __restrict__ ebuf, unsigned* __restrict__ menc) {
    int t = blockIdx.x * blockDim.x + threadIdx.x;
    if (t >= EE * HH) return;
    int e = t >> 2, h = t & 3;
    int s = src[e], d = dst[e];
    const float* ps = hs + (long)s * DD + h * CC;
    const float* pd = hd + (long)d * DD + h * CC;
    const float* pa = avec + h * CC;
    float sum = 0.f;
#pragma unroll 4
    for (int c = 0; c < CC; ++c) {
        float v = pd[c] + ps[c];
        v = (v > 0.f) ? v : 0.2f * v;     // leaky_relu(0.2)
        sum += pa[c] * v;
    }
    ebuf[t] = sum;
    atomicMax(&menc[d * HH + h], fenc(sum));
}

__global__ void edge_softmax(const int* __restrict__ dst,
                             const unsigned* __restrict__ menc,
                             float* __restrict__ ebuf, float* __restrict__ den) {
    int t = blockIdx.x * blockDim.x + threadIdx.x;
    if (t >= EE * HH) return;
    int e = t >> 2, h = t & 3;
    int d = dst[e];
    float m = fdec(menc[d * HH + h]);
    if (!isfinite(m)) m = 0.f;
    float p = __expf(ebuf[t] - m);
    ebuf[t] = p;
    atomicAdd(&den[d * HH + h], p);
}

__global__ void edge_message(const float* __restrict__ hs,
                             const int* __restrict__ src, const int* __restrict__ dst,
                             const float* __restrict__ ebuf, const float* __restrict__ den,
                             const float* __restrict__ emask, float* __restrict__ hout) {
    int t = blockIdx.x * blockDim.x + threadIdx.x;
    if (t >= EE * HH) return;
    int e = t >> 2, h = t & 3;
    int s = src[e], d = dst[e];
    float alpha = ebuf[t] / (den[d * HH + h] + 1e-16f);
    float w = alpha * emask[e];
    const float* ps = hs + (long)s * DD + h * CC;
    float* po = hout + (long)d * DD + h * CC;
#pragma unroll 4
    for (int c = 0; c < CC; ++c) atomicAdd(&po[c], w * ps[c]);
}

// -------- pooling: S = softmax(x_last @ W_assign); x_pool += S^T x_last -----
static constexpr int POOL_NODES = 128;
__global__ __launch_bounds__(256)
void pool_kernel(const float* __restrict__ xl, const float* __restrict__ Wassign,
                 float* __restrict__ xpool, int nrows) {
    __shared__ float xrow[DD];
    __shared__ float sv[KK];
    const int tid = threadIdx.x;
    float accp[KK];
#pragma unroll
    for (int k = 0; k < KK; ++k) accp[k] = 0.f;

    int n0 = blockIdx.x * POOL_NODES;
    int n1 = min(n0 + POOL_NODES, nrows);
    for (int n = n0; n < n1; ++n) {
        xrow[tid] = xl[(long)n * DD + tid];
        __syncthreads();
        if (tid < KK) {
            float s = 0.f;
            for (int d = 0; d < DD; ++d) s += xrow[d] * Wassign[d * KK + tid];
            sv[tid] = s;
        }
        __syncthreads();
        float mx = -3.0e38f;
#pragma unroll
        for (int k = 0; k < KK; ++k) mx = fmaxf(mx, sv[k]);
        float dsum = 0.f;
#pragma unroll
        for (int k = 0; k < KK; ++k) dsum += __expf(sv[k] - mx);
        float xv = xrow[tid];
        float inv = 1.f / dsum;
#pragma unroll
        for (int k = 0; k < KK; ++k) accp[k] += (__expf(sv[k] - mx) * inv) * xv;
        __syncthreads();
    }
#pragma unroll
    for (int k = 0; k < KK; ++k) atomicAdd(&xpool[k * DD + tid], accp[k]);
}

// ---------------- edge_attr means + classifier head ----------------
__global__ __launch_bounds__(256)
void edge_mean(const float* __restrict__ ea, float* __restrict__ agg) {
    __shared__ float red[256];
    int t = blockIdx.y;
    long i = (long)blockIdx.x * 256 + threadIdx.x;
    red[threadIdx.x] = (i < EE) ? ea[(long)t * EE + i] : 0.f;
    __syncthreads();
    for (int s = 128; s > 0; s >>= 1) {
        if (threadIdx.x < s) red[threadIdx.x] += red[threadIdx.x + s];
        __syncthreads();
    }
    if (threadIdx.x == 0) atomicAdd(&agg[t], red[0]);
}

__global__ void final_head(const float* __restrict__ xpool, const float* __restrict__ agg,
                           const float* __restrict__ Wcls, const float* __restrict__ bcls,
                           float* __restrict__ out) {
    int k = threadIdx.x;
    if (k >= KK) return;
    float acc = bcls[0];
    for (int d = 0; d < DD; ++d) acc += xpool[k * DD + d] * Wcls[d];
    const float invE = 1.f / (float)EE;
#pragma unroll
    for (int t = 0; t < T; ++t) acc += (agg[t] * invE) * Wcls[DD + t];
    out[k] = 1.f / (1.f + __expf(-acc));
}

// ---------------- host launcher ----------------
extern "C" void kernel_launch(void* const* d_in, const int* in_sizes, int n_in,
                              void* d_out, int out_size, void* d_ws, size_t ws_size,
                              hipStream_t stream) {
    const float* x_pkg  = (const float*)d_in[0];
    const float* x_dst  = (const float*)d_in[1];
    const int*   eidx   = (const int*)d_in[2];
    const float* eattr  = (const float*)d_in[3];
    const float* nmask  = (const float*)d_in[4];
    const float* emask  = (const float*)d_in[5];
    const float* W1s    = (const float*)d_in[6];
    const float* W1d    = (const float*)d_in[7];
    const float* a1     = (const float*)d_in[8];
    const float* W2s    = (const float*)d_in[9];
    const float* W2d    = (const float*)d_in[10];
    const float* a2     = (const float*)d_in[11];
    const float* Wasn   = (const float*)d_in[12];
    const float* Wcls   = (const float*)d_in[13];
    const float* bcls   = (const float*)d_in[14];
    float* out = (float*)d_out;

    // t = T-1 slices (only live path)
    const int tl = T - 1;
    const float* x_dst5 = x_dst + (long)tl * NN * DD;
    const int* src5 = eidx + (long)tl * 2 * EE;
    const int* dst5 = src5 + EE;
    const float* W1s5 = W1s + (long)tl * DD * DD;
    const float* W1d5 = W1d + (long)tl * DD * DD;
    const float* W2s5 = W2s + (long)tl * DD * DD;
    const float* W2d5 = W2d + (long)tl * DD * DD;
    const float* a1_5 = a1 + tl * DD;
    const float* a2_5 = a2 + tl * DD;

    // workspace carve-up
    char* base = (char*)d_ws;
    size_t off = 0;
    auto carve = [&](size_t bytes) { char* p = base + off; off += (bytes + 255) & ~(size_t)255; return p; };
    _Float16* xs1  = (_Float16*)carve((size_t)NN * DD * 2);
    _Float16* xdb  = (_Float16*)carve((size_t)NN * DD * 2);   // xd1, later relu(h1)
    _Float16* xs2  = (_Float16*)carve((size_t)NN * DD * 2);
    _Float16* wt1s = (_Float16*)carve((size_t)DD * DD * 2);
    _Float16* wt1d = (_Float16*)carve((size_t)DD * DD * 2);
    _Float16* wt2s = (_Float16*)carve((size_t)DD * DD * 2);
    _Float16* wt2d = (_Float16*)carve((size_t)DD * DD * 2);
    float*    hs   = (float*)carve((size_t)NN * DD * 4);
    float*    hd   = (float*)carve((size_t)NN * DD * 4);
    float*    hacc = (float*)carve((size_t)NN * DD * 4);      // h1, then h2 (= x_last)
    float*    ebuf = (float*)carve((size_t)EE * HH * 4);
    unsigned* menc = (unsigned*)carve((size_t)NN * HH * 4);
    float*    den  = (float*)carve((size_t)NN * HH * 4);
    float*    xpool= (float*)carve((size_t)KK * DD * 4);
    float*    agg  = (float*)carve(256);
    (void)ws_size; (void)in_sizes; (void)n_in; (void)out_size;

    const int ND = NN * DD;
    dim3 g256((ND + 255) / 256);
    dim3 gW((DD * DD + 255) / 256);
    dim3 gGemm((NN + 127) / 128, 2);
    dim3 gEdge((EE * HH + 255) / 256);

    // --- prep: masked/activated f16 inputs + transposed f16 weights ---
    prep_inputs<<<g256, 256, 0, stream>>>(x_pkg, x_dst5, nmask, xs1, xdb, xs2, ND);
    convert_w<<<gW, 256, 0, stream>>>(W1s5, wt1s);
    convert_w<<<gW, 256, 0, stream>>>(W1d5, wt1d);
    convert_w<<<gW, 256, 0, stream>>>(W2s5, wt2s);
    convert_w<<<gW, 256, 0, stream>>>(W2d5, wt2d);

    // --- layer 1 GEMMs (WMMA + async double-buffered staging) ---
    gemm_wmma_f16<<<gGemm, 256, GEMM_LDS, stream>>>(xs1, wt1s, hs, NN);
    gemm_wmma_f16<<<gGemm, 256, GEMM_LDS, stream>>>(xdb, wt1d, hd, NN);

    // --- layer 1 edge softmax + aggregation ---
    zero_u32<<<(NN * HH + 255) / 256, 256, 0, stream>>>(menc, (long)NN * HH);
    zero_u32<<<(NN * HH + 255) / 256, 256, 0, stream>>>((unsigned*)den, (long)NN * HH);
    zero_u32<<<g256, 256, 0, stream>>>((unsigned*)hacc, (long)ND);
    edge_scores<<<gEdge, 256, 0, stream>>>(hs, hd, src5, dst5, a1_5, ebuf, menc);
    edge_softmax<<<gEdge, 256, 0, stream>>>(dst5, menc, ebuf, den);
    edge_message<<<gEdge, 256, 0, stream>>>(hs, src5, dst5, ebuf, den, emask, hacc);

    // --- layer 2: xd2 = relu(h1) in f16; GEMMs reuse hs/hd ---
    relu_to_f16<<<g256, 256, 0, stream>>>(hacc, xdb, ND);
    gemm_wmma_f16<<<gGemm, 256, GEMM_LDS, stream>>>(xs2, wt2s, hs, NN);
    gemm_wmma_f16<<<gGemm, 256, GEMM_LDS, stream>>>(xdb, wt2d, hd, NN);

    zero_u32<<<(NN * HH + 255) / 256, 256, 0, stream>>>(menc, (long)NN * HH);
    zero_u32<<<(NN * HH + 255) / 256, 256, 0, stream>>>((unsigned*)den, (long)NN * HH);
    zero_u32<<<g256, 256, 0, stream>>>((unsigned*)hacc, (long)ND);
    edge_scores<<<gEdge, 256, 0, stream>>>(hs, hd, src5, dst5, a2_5, ebuf, menc);
    edge_softmax<<<gEdge, 256, 0, stream>>>(dst5, menc, ebuf, den);
    edge_message<<<gEdge, 256, 0, stream>>>(hs, src5, dst5, ebuf, den, emask, hacc);

    // --- pooling + edge means + classifier ---
    zero_u32<<<(KK * DD + 255) / 256, 256, 0, stream>>>((unsigned*)xpool, KK * DD);
    zero_u32<<<1, 64, 0, stream>>>((unsigned*)agg, T);
    pool_kernel<<<(NN + POOL_NODES - 1) / POOL_NODES, 256, 0, stream>>>(hacc, Wasn, xpool, NN);
    edge_mean<<<dim3((EE + 255) / 256, T), 256, 0, stream>>>(eattr, agg);
    final_head<<<1, 64, 0, stream>>>(xpool, agg, Wcls, bcls, out);
}
